// AttentionModuleV1_24068996726851
// MI455X (gfx1250) — matrix operand order repaired and checked by reading, not compiled, not run on hardware
//
#include <hip/hip_runtime.h>
#include <hip/hip_bf16.h>

// ---------------------------------------------------------------------------
// MI455X (gfx1250) fused implementation.
//
// Math collapse (exact): the reference broadcasts features along L=16, so the
// softmax is over identical logits -> uniform 1/16 -> attention is identity.
//   P   = relu(F1 @ Wk^T)
//   H   = relu(bn1(F2 @ Wv1^T))          (BN folded into weights+bias)
//   V   = relu(bn2(H @ Wv2^T))
//   out = V + P * sigmoid(P @ Wv_coef^T)
// Four 16384x256x256 GEMMs -> bf16 WMMA (v_wmma_f32_16x16x32_bf16), fp32 acc.
//
// Each wave owns 32 rows (two 16-row M-tiles): every B (weight) fragment load
// feeds two WMMAs (1 vmem : 1 wmma). Column-tile loop unrolled x2 so the
// scheduler sees four independent WMMA chains and can keep more b128 loads
// in flight across tile boundaries.
// ---------------------------------------------------------------------------

typedef __bf16 bf16_t;
typedef __attribute__((ext_vector_type(16))) __bf16 v16bf;
typedef __attribute__((ext_vector_type(8)))  __bf16 v8bf;
typedef __attribute__((ext_vector_type(8)))  float  v8f;
typedef __attribute__((ext_vector_type(4)))  float  v4f;

#define NROWS 16384
#define CIN   256
#define COUT  256
#define RPB   128      // rows per block (4 waves x 32 rows)
#define BN_EPS 1e-5f

// ---------------- prep: fold BN into bf16 weights ---------------------------
__global__ __launch_bounds__(256) void prep_kernel(
    const float* __restrict__ Wk,  const float* __restrict__ Wv1,
    const float* __restrict__ Wv2, const float* __restrict__ Wvc,
    const float* __restrict__ g1, const float* __restrict__ b1,
    const float* __restrict__ m1, const float* __restrict__ v1,
    const float* __restrict__ g2, const float* __restrict__ b2,
    const float* __restrict__ m2, const float* __restrict__ v2,
    bf16_t* __restrict__ oWk,  bf16_t* __restrict__ oWv1,
    bf16_t* __restrict__ oWv2, bf16_t* __restrict__ oWvc,
    float* __restrict__ bias0, float* __restrict__ bias1,
    float* __restrict__ bias2)
{
    const int mat = blockIdx.x >> 8;     // 0..3
    const int row = blockIdx.x & 255;
    const int col = threadIdx.x;
    const int idx = row * 256 + col;
    if (mat == 0) {
        oWk[idx] = (bf16_t)Wk[idx];
        if (col == 0) bias0[row] = 0.0f;          // zero bias (branch-free loop)
    } else if (mat == 1) {
        const float inv = g1[row] * rsqrtf(v1[row] + BN_EPS);
        oWv1[idx] = (bf16_t)(Wv1[idx] * inv);
        if (col == 0) bias1[row] = b1[row] - m1[row] * inv;
    } else if (mat == 2) {
        const float inv = g2[row] * rsqrtf(v2[row] + BN_EPS);
        oWv2[idx] = (bf16_t)(Wv2[idx] * inv);
        if (col == 0) bias2[row] = b2[row] - m2[row] * inv;
    } else {
        oWvc[idx] = (bf16_t)Wvc[idx];
    }
}

// ---------------- fragment helpers -----------------------------------------
__device__ __forceinline__ v8f zero8() {
    v8f z;
#pragma unroll
    for (int i = 0; i < 8; ++i) z[i] = 0.0f;
    return z;
}

__device__ __forceinline__ v8f wmma_bf16(v16bf a, v16bf b, v8f c) {
    // (neg_a, A, neg_b, B, c_mod, C, reuse_a, reuse_b)
    return __builtin_amdgcn_wmma_f32_16x16x32_bf16(false, a, false, b,
                                                   (short)0, c, false, false);
}

// A fragment (16x32 bf16, ISA layout): lane row = lane&15; this lane holds
// K = kb..kb+7 (elems 0-7) and kb+16..kb+23 (elems 8-15), kb = kc*32 + (lane>=16?8:0).
__device__ __forceinline__ v16bf make_a_from_f32(const float* __restrict__ rowp, int kb) {
    const v4f* p = (const v4f*)(rowp + kb);
    v4f q0 = p[0], q1 = p[1], q2 = p[4], q3 = p[5];
    float t[16] = {q0.x, q0.y, q0.z, q0.w, q1.x, q1.y, q1.z, q1.w,
                   q2.x, q2.y, q2.z, q2.w, q3.x, q3.y, q3.z, q3.w};
    v16bf r;
#pragma unroll
    for (int i = 0; i < 16; ++i) r[i] = (bf16_t)t[i];
    return r;
}

// Same A layout, sourced from bf16 LDS rows.
__device__ __forceinline__ v16bf make_a_from_bf16(const bf16_t* __restrict__ rowp, int kb) {
    const v8bf* q0 = (const v8bf*)(rowp + kb);
    const v8bf* q1 = (const v8bf*)(rowp + kb + 16);
    return __builtin_shufflevector(q0[0], q1[0],
                                   0, 1, 2, 3, 4, 5, 6, 7,
                                   8, 9, 10, 11, 12, 13, 14, 15);
}

// B fragment (32x16 bf16): lane col = lane&15; elems 0-7 hold K = k0..k0+7,
// elems 8-15 hold K = k0+16..k0+23, k0 = kc*32 + (lane>=16?8:0).
__device__ __forceinline__ v16bf load_b(const bf16_t* __restrict__ p) {
    const v8bf* q = (const v8bf*)p;
    return __builtin_shufflevector(q[0], q[2],
                                   0, 1, 2, 3, 4, 5, 6, 7,
                                   8, 9, 10, 11, 12, 13, 14, 15);
}

// ---------------- fused kernel ----------------------------------------------
__global__ __launch_bounds__(128) void fused_kernel(
    const float* __restrict__ F1, const float* __restrict__ F2,
    const bf16_t* __restrict__ Wk,  const bf16_t* __restrict__ Wv1,
    const bf16_t* __restrict__ Wv2, const bf16_t* __restrict__ Wvc,
    const float* __restrict__ bias0, const float* __restrict__ bias1,
    const float* __restrict__ bias2, float* __restrict__ out)
{
    extern __shared__ char smem[];
    bf16_t* sP = (bf16_t*)smem;                     // [128][256] relu(F1@Wk^T)
    bf16_t* sH = (bf16_t*)(smem + RPB * COUT * 2);  // [128][256] H, then recycled as V

    const int tid  = threadIdx.x;
    const int wave = tid >> 5;
    const int lane = tid & 31;
    const int lhi  = lane >> 4;            // 0 | 1
    const int lmod = lane & 15;
    const int kA   = lhi * 8;              // A-fragment K sub-offset
    const int kB   = lhi * 8;              // B-fragment K sub-offset
    const int rowT0 = wave * 32 + lmod;    // block-local A row, tile 0
    const int rowT1 = rowT0 + 16;          // tile 1
    const size_t gRow0 = (size_t)blockIdx.x * RPB + rowT0;
    const size_t gRow1 = gRow0 + 16;
    const int mBase0 = wave * 32 + lhi * 8;   // D-layout M base, tile 0
    const int mBase1 = mBase0 + 16;           // tile 1

    // ---- phases 0/1: P = relu(F1@Wk^T) -> sP ; H = relu(bn1(F2@Wv1^T)) -> sH
    for (int ph = 0; ph < 2; ++ph) {
        const float*  F    = ph ? F2    : F1;
        const bf16_t* W    = ph ? Wv1   : Wk;
        const float*  bias = ph ? bias1 : bias0;
        bf16_t*       dst  = ph ? sH    : sP;

        v16bf a0[8], a1[8];
#pragma unroll
        for (int kc = 0; kc < 8; ++kc) {
            a0[kc] = make_a_from_f32(F + gRow0 * CIN, kc * 32 + kA);
            a1[kc] = make_a_from_f32(F + gRow1 * CIN, kc * 32 + kA);
        }
#pragma unroll 2
        for (int nc = 0; nc < 16; ++nc) {
            const int ncol = nc * 16 + lmod;         // output channel
            v16bf bb[8];
#pragma unroll
            for (int kc = 0; kc < 8; ++kc)           // one big load clause
                bb[kc] = load_b(W + ncol * CIN + kc * 32 + kB);
            const float bv = bias[ncol];
            v8f acc0 = zero8(), acc1 = zero8();
#pragma unroll
            for (int kc = 0; kc < 8; ++kc) {         // drain with partial waits
                acc0 = wmma_bf16(a0[kc], bb[kc], acc0);
                acc1 = wmma_bf16(a1[kc], bb[kc], acc1);
            }
#pragma unroll
            for (int v = 0; v < 8; ++v) {
                dst[(mBase0 + v) * COUT + ncol] = (bf16_t)fmaxf(acc0[v] + bv, 0.0f);
                dst[(mBase1 + v) * COUT + ncol] = (bf16_t)fmaxf(acc1[v] + bv, 0.0f);
            }
        }
    }

    __syncthreads();

    // ---- phase 2: V = relu(bn2(H@Wv2^T)); recycle sH to hold bf16 V ----
    {
        v16bf h0[8], h1[8];
        const bf16_t* r0 = sH + rowT0 * COUT;
        const bf16_t* r1 = sH + rowT1 * COUT;
#pragma unroll
        for (int kc = 0; kc < 8; ++kc) {
            h0[kc] = make_a_from_bf16(r0, kc * 32 + kA);
            h1[kc] = make_a_from_bf16(r1, kc * 32 + kA);
        }
#pragma unroll 2
        for (int nc = 0; nc < 16; ++nc) {
            const int ncol = nc * 16 + lmod;
            v16bf bb[8];
#pragma unroll
            for (int kc = 0; kc < 8; ++kc)
                bb[kc] = load_b(Wv2 + ncol * COUT + kc * 32 + kB);
            const float bv = bias2[ncol];
            v8f acc0 = zero8(), acc1 = zero8();
#pragma unroll
            for (int kc = 0; kc < 8; ++kc) {
                acc0 = wmma_bf16(h0[kc], bb[kc], acc0);
                acc1 = wmma_bf16(h1[kc], bb[kc], acc1);
            }
#pragma unroll
            for (int v = 0; v < 8; ++v) {   // wave-private rows: safe overwrite
                sH[(mBase0 + v) * COUT + ncol] = (bf16_t)fmaxf(acc0[v] + bv, 0.0f);
                sH[(mBase1 + v) * COUT + ncol] = (bf16_t)fmaxf(acc1[v] + bv, 0.0f);
            }
        }
    }

    __syncthreads();

    // ---- phase 3: G = P@Wvc^T; out = V + P*sigmoid(G) ----
    {
        v16bf p0[8], p1[8];
        const bf16_t* r0 = sP + rowT0 * COUT;
        const bf16_t* r1 = sP + rowT1 * COUT;
#pragma unroll
        for (int kc = 0; kc < 8; ++kc) {
            p0[kc] = make_a_from_bf16(r0, kc * 32 + kA);
            p1[kc] = make_a_from_bf16(r1, kc * 32 + kA);
        }
#pragma unroll 2
        for (int nc = 0; nc < 16; ++nc) {
            const int ncol = nc * 16 + lmod;
            v16bf bb[8];
#pragma unroll
            for (int kc = 0; kc < 8; ++kc)
                bb[kc] = load_b(Wvc + ncol * COUT + kc * 32 + kB);
            v8f acc0 = zero8(), acc1 = zero8();
#pragma unroll
            for (int kc = 0; kc < 8; ++kc) {
                acc0 = wmma_bf16(p0[kc], bb[kc], acc0);
                acc1 = wmma_bf16(p1[kc], bb[kc], acc1);
            }
#pragma unroll
            for (int v = 0; v < 8; ++v) {
                const int m0 = mBase0 + v, m1 = mBase1 + v;
                const float g0 = 1.0f / (1.0f + __expf(-acc0[v]));
                const float g1 = 1.0f / (1.0f + __expf(-acc1[v]));
                const float pv0 = (float)sP[m0 * COUT + ncol];
                const float pv1 = (float)sP[m1 * COUT + ncol];
                const float vv0 = (float)sH[m0 * COUT + ncol];
                const float vv1 = (float)sH[m1 * COUT + ncol];
                const size_t gr0 = (size_t)blockIdx.x * RPB + m0;
                const size_t gr1 = (size_t)blockIdx.x * RPB + m1;
                out[gr0 * COUT + ncol] = vv0 + pv0 * g0;
                out[gr1 * COUT + ncol] = vv1 + pv1 * g1;
            }
        }
    }
}

// ---------------- launch -----------------------------------------------------
extern "C" void kernel_launch(void* const* d_in, const int* in_sizes, int n_in,
                              void* d_out, int out_size, void* d_ws, size_t ws_size,
                              hipStream_t stream) {
    (void)in_sizes; (void)n_in; (void)out_size; (void)ws_size;
    const float* features  = (const float*)d_in[1];
    const float* features2 = (const float*)d_in[2];
    const float* Wk   = (const float*)d_in[3];
    const float* Wv1  = (const float*)d_in[4];
    const float* bn1g = (const float*)d_in[5];
    const float* bn1b = (const float*)d_in[6];
    const float* bn1m = (const float*)d_in[7];
    const float* bn1v = (const float*)d_in[8];
    const float* Wv2  = (const float*)d_in[9];
    const float* bn2g = (const float*)d_in[10];
    const float* bn2b = (const float*)d_in[11];
    const float* bn2m = (const float*)d_in[12];
    const float* bn2v = (const float*)d_in[13];
    const float* Wvc  = (const float*)d_in[18];   // Wv_coef (others are dead)

    char* ws = (char*)d_ws;                       // needs ~516 KB
    bf16_t* wsWk  = (bf16_t*)(ws);
    bf16_t* wsWv1 = (bf16_t*)(ws + 131072);
    bf16_t* wsWv2 = (bf16_t*)(ws + 262144);
    bf16_t* wsWvc = (bf16_t*)(ws + 393216);
    float*  bias0 = (float*)(ws + 524288);
    float*  bias1 = (float*)(ws + 525312);
    float*  bias2 = (float*)(ws + 526336);

    prep_kernel<<<1024, 256, 0, stream>>>(Wk, Wv1, Wv2, Wvc,
                                          bn1g, bn1b, bn1m, bn1v,
                                          bn2g, bn2b, bn2m, bn2v,
                                          wsWk, wsWv1, wsWv2, wsWvc,
                                          bias0, bias1, bias2);

    const int smem_bytes = 2 * RPB * COUT * 2;    // 128 KB (<= 320 KB/WGP)
    (void)hipFuncSetAttribute((const void*)fused_kernel,
                              hipFuncAttributeMaxDynamicSharedMemorySize, smem_bytes);
    fused_kernel<<<NROWS / RPB, 128, smem_bytes, stream>>>(
        features, features2, wsWk, wsWv1, wsWv2, wsWvc,
        bias0, bias1, bias2, (float*)d_out);
}
